// personaGAT_16174846836805
// MI455X (gfx1250) — matrix operand order repaired and checked by reading, not compiled, not run on hardware
//
#include <hip/hip_runtime.h>

typedef __attribute__((ext_vector_type(16))) __bf16 v16bf;
typedef __attribute__((ext_vector_type(8)))  float  v8f;

#define HEADS 4
#define HD 32
#define CDIM 128
#define FRAGS_PER_LAYER 40   // 4 gate + 4 persona-scalar + 32 Wx frags
#define FRAG_ELEMS 512       // 32 lanes * 16 bf16
#define FRAG_BYTES_PER_LAYER (FRAGS_PER_LAYER * FRAG_ELEMS * 2)  // 40 KB

// ---------- helpers ----------
// native fptrunc: lets the backend pick v_cvt_pk_bf16_f32 on gfx1250
__device__ inline __bf16 f2bf(float f) { return (__bf16)f; }

// order-preserving float<->uint encoding for atomicMax on signed floats
__device__ inline unsigned encf(float f) {
  unsigned u = __builtin_bit_cast(unsigned, f);
  return (u & 0x80000000u) ? ~u : (u | 0x80000000u);
}
__device__ inline float decf(unsigned e) {
  unsigned u = (e & 0x80000000u) ? (e & 0x7FFFFFFFu) : ~e;
  return __builtin_bit_cast(float, u);
}

// A-fragment (16x32 bf16, wave32): lane l holds row m=l&15; with g=l>>4,
// elements 0..7 -> K = g*8 + c*32 + 0..7, elements 8..15 -> +16.
__device__ inline v16bf load_a(const float* p) {
  float4 p0 = *(const float4*)(p);
  float4 p1 = *(const float4*)(p + 4);
  float4 p2 = *(const float4*)(p + 16);
  float4 p3 = *(const float4*)(p + 20);
  v16bf a;
  a[0]=f2bf(p0.x);  a[1]=f2bf(p0.y);  a[2]=f2bf(p0.z);  a[3]=f2bf(p0.w);
  a[4]=f2bf(p1.x);  a[5]=f2bf(p1.y);  a[6]=f2bf(p1.z);  a[7]=f2bf(p1.w);
  a[8]=f2bf(p2.x);  a[9]=f2bf(p2.y);  a[10]=f2bf(p2.z); a[11]=f2bf(p2.w);
  a[12]=f2bf(p3.x); a[13]=f2bf(p3.y); a[14]=f2bf(p3.z); a[15]=f2bf(p3.w);
  return a;
}

__device__ inline v8f wmma_bf16(v16bf a, v16bf b, v8f c) {
  return __builtin_amdgcn_wmma_f32_16x16x32_bf16(false, a, false, b, (short)0, c,
                                                 false, false);
}

// ---------- prep: fold Wp@att_W and pre-swizzle B fragments to bf16 ----------
// B-frag layout (32x16 bf16, wave32): lane l holds col n=l&15, elements e hold
// K = (e&7) + (l>>4)*8 + (e>>3)*16. Stored lane-contiguous: frags[f*512 + lane*16 + e].
__global__ void k_prep(const float* __restrict__ gate_W, const float* __restrict__ Wp,
                       const float* __restrict__ Wx, const float* __restrict__ att_W,
                       __bf16* __restrict__ frags) {
  int idx = blockIdx.x * blockDim.x + threadIdx.x;
  if (idx >= 2 * FRAGS_PER_LAYER * FRAG_ELEMS) return;
  int f    = idx >> 9;
  int el   = idx & 511;
  int lane = el >> 4;
  int e    = el & 15;
  int kloc = (e & 7) + ((lane >> 4) << 3) + ((e >> 3) << 4);
  int n    = lane & 15;
  int l = f / FRAGS_PER_LAYER;
  int r = f % FRAGS_PER_LAYER;
  float v = 0.0f;
  if (r < 4) {                    // gate B: cols 0..7 = gate_W[h = n&3] (duplicated)
    int k = r * 32 + kloc;
    if (n < 8) { int h = n & 3; v = gate_W[(size_t)(l*HEADS + h)*CDIM + k]; }
  } else if (r < 8) {             // persona B: cols 0..3 = Wp@aw_lo, 4..7 = Wp@aw_hi
    int c = r - 4; int k = c * 32 + kloc;
    if (n < 8) {
      int h = n & 3;
      const float* aw = att_W + (size_t)(l*HEADS + h)*(2*HD) + ((n < 4) ? 0 : HD);
      const float* wp = Wp + (size_t)((l*HEADS + h)*CDIM + k) * HD;
      float s = 0.f;
      for (int dd = 0; dd < HD; ++dd) s += wp[dd] * aw[dd];
      v = s;
    }
  } else {                        // Wx B frags: (h, coltile t, kchunk c)
    int r2 = r - 8; int c = r2 & 3; int ht = r2 >> 2; int h = ht >> 1; int t = ht & 1;
    int k = c * 32 + kloc;
    v = Wx[(size_t)((l*HEADS + h)*CDIM + k) * HD + t*16 + n];
  }
  frags[idx] = f2bf(v);
}

// ---------- K1: node transform via WMMA ----------
__global__ __launch_bounds__(256) void k1_node(
    const float* __restrict__ xin, const float* __restrict__ persona,
    const __bf16* __restrict__ frags, const float* __restrict__ gate_b,
    float* __restrict__ asrc, float* __restrict__ adst, float* __restrict__ xf,
    int N, int layer) {
  __shared__ __align__(32) __bf16 sfr[FRAGS_PER_LAYER * FRAG_ELEMS];  // 40 KB
  {
    // CDNA5 async global->LDS staging of the pre-swizzled weight fragments.
    // Per-lane 16B transfers; tracked by ASYNCcnt (no VGPR data path).
    const unsigned char* gbase =
        (const unsigned char*)(frags + (size_t)layer * FRAGS_PER_LAYER * FRAG_ELEMS);
    unsigned ldsbase = (unsigned)(unsigned long long)(void*)&sfr[0];
    for (int i = threadIdx.x; i < FRAG_BYTES_PER_LAYER / 16; i += blockDim.x) {
      unsigned long long ga = (unsigned long long)(gbase + (size_t)i * 16);
      unsigned la = ldsbase + (unsigned)i * 16;
      asm volatile("global_load_async_to_lds_b128 %0, %1, off"
                   :: "v"(la), "v"(ga) : "memory");
    }
    asm volatile("s_wait_asynccnt 0x0" ::: "memory");
  }
  __syncthreads();

  int wave = threadIdx.x >> 5;
  int lane = threadIdx.x & 31;
  int tiles = (N + 15) >> 4;
  int tile = blockIdx.x * 8 + wave;
  if (tile >= tiles) return;           // wave-uniform: EXEC all-ones for WMMA
  int g16 = lane >> 4;
  int m   = lane & 15;
  int node = tile * 16 + m;
  int node_c = node < N ? node : N - 1;

  v16bf xa[4], pa[4];
  #pragma unroll
  for (int c = 0; c < 4; ++c) {
    xa[c] = load_a(xin     + (size_t)node_c * CDIM + c * 32 + g16 * 8);
    pa[c] = load_a(persona + (size_t)node_c * CDIM + c * 32 + g16 * 8);
  }

  // gate (cols 0..7 duplicated) and persona attention scalars
  v8f dg = {0,0,0,0,0,0,0,0};
  v8f dp = {0,0,0,0,0,0,0,0};
  #pragma unroll
  for (int c = 0; c < 4; ++c) {
    v16bf bg = *(const v16bf*)(sfr + c * FRAG_ELEMS + lane * 16);
    dg = wmma_bf16(xa[c], bg, dg);
  }
  #pragma unroll
  for (int c = 0; c < 4; ++c) {
    v16bf bp = *(const v16bf*)(sfr + (4 + c) * FRAG_ELEMS + lane * 16);
    dp = wmma_bf16(pa[c], bp, dp);
  }
  // D layout: lane col n=l&15, element e = row m = e + g16*8
  if (m < 8) {
    int h = m & 3;
    float gb = gate_b[layer * HEADS + h];
    float* outp = ((m < 4) ? asrc : adst) + (size_t)h * N + tile * 16 + g16 * 8;
    int base = tile * 16 + g16 * 8;
    #pragma unroll
    for (int e = 0; e < 8; ++e)
      if (base + e < N) outp[e] = (dg[e] + gb) * dp[e];
  }

  // xf = x @ Wx[h]  -> [H][N][32]
  #pragma unroll
  for (int h = 0; h < HEADS; ++h) {
    #pragma unroll
    for (int t = 0; t < 2; ++t) {
      v8f acc = {0,0,0,0,0,0,0,0};
      #pragma unroll
      for (int c = 0; c < 4; ++c) {
        v16bf b = *(const v16bf*)(sfr + (8 + ((h*2+t)*4 + c)) * FRAG_ELEMS + lane * 16);
        acc = wmma_bf16(xa[c], b, acc);
      }
      int rowbase = tile * 16 + g16 * 8;
      float* dst = xf + ((size_t)h * N + rowbase) * HD + t * 16 + m;
      #pragma unroll
      for (int e = 0; e < 8; ++e)
        if (rowbase + e < N) dst[(size_t)e * HD] = acc[e];
    }
  }
}

// ---------- K2: segment max of leaky-relu scores ----------
__global__ void k2_max(const int* __restrict__ srcv, const int* __restrict__ dstv,
                       const float* __restrict__ asrc, const float* __restrict__ adst,
                       unsigned* __restrict__ mEnc, int N, int E) {
  int idx = blockIdx.x * blockDim.x + threadIdx.x;
  int total = (E + N) * HEADS;
  if (idx >= total) return;
  int h = idx & 3;
  int eidx = idx >> 2;
  int s, d;
  if (eidx < E) { s = srcv[eidx]; d = dstv[eidx]; } else { s = d = eidx - E; }
  float e = asrc[(size_t)h * N + s] + adst[(size_t)h * N + d];
  e = e > 0.0f ? e : 0.2f * e;
  atomicMax(mEnc + (size_t)h * N + d, encf(e));
}

// ---------- K3: fused exp + denominator + weighted scatter-add ----------
__global__ void k3_acc(const int* __restrict__ srcv, const int* __restrict__ dstv,
                       const float* __restrict__ asrc, const float* __restrict__ adst,
                       const unsigned* __restrict__ mEnc, float* __restrict__ den,
                       const float* __restrict__ xf, float* __restrict__ num,
                       int N, int E) {
  long idx = (long)blockIdx.x * blockDim.x + threadIdx.x;
  long total = (long)(E + N) * HEADS * 8;
  if (idx >= total) return;
  int  q    = (int)(idx & 7);        // 8 threads per (edge,head): 4 floats each
  long grp  = idx >> 3;
  int  h    = (int)(grp & 3);
  long eidx = grp >> 2;
  int s, d;
  if (eidx < E) { s = srcv[eidx]; d = dstv[eidx]; } else { s = d = (int)(eidx - E); }
  float e = asrc[(size_t)h * N + s] + adst[(size_t)h * N + d];
  e = e > 0.0f ? e : 0.2f * e;
  float mx = decf(mEnc[(size_t)h * N + d]);
  float ex = __expf(e - mx);
  if (q == 0) atomicAdd(den + (size_t)h * N + d, ex);
  const float* xs = xf + ((size_t)h * N + s) * HD + q * 4;
  float*       nd = num + ((size_t)h * N + d) * HD + q * 4;
  float4 v = *(const float4*)xs;
  atomicAdd(nd + 0, ex * v.x);
  atomicAdd(nd + 1, ex * v.y);
  atomicAdd(nd + 2, ex * v.z);
  atomicAdd(nd + 3, ex * v.w);
}

// ---------- K4: normalize + bias + ELU + residual ----------
__global__ void k4_fin(const float* __restrict__ xin, const float* __restrict__ num,
                       const float* __restrict__ den, const float* __restrict__ bias,
                       float* __restrict__ out, int N, int layer) {
  long idx = (long)blockIdx.x * blockDim.x + threadIdx.x;
  if (idx >= (long)N * CDIM) return;
  int n = (int)(idx >> 7);
  int c = (int)(idx & 127);
  int h = c >> 5; int dd = c & 31;
  float agg = num[((size_t)h * N + n) * HD + dd] / (den[(size_t)h * N + n] + 1e-16f);
  float o = agg + bias[layer * CDIM + c];
  o = o > 0.0f ? o : (__expf(o) - 1.0f);           // ELU
  out[idx] = xin[idx] + o;
}

__global__ void k_fill0(unsigned* __restrict__ p, long nwords) {
  long i = (long)blockIdx.x * blockDim.x + threadIdx.x;
  long stride = (long)gridDim.x * blockDim.x;
  for (; i < nwords; i += stride) p[i] = 0u;
}

// ---------- host ----------
extern "C" void kernel_launch(void* const* d_in, const int* in_sizes, int n_in,
                              void* d_out, int out_size, void* d_ws, size_t ws_size,
                              hipStream_t stream) {
  const float* x0      = (const float*)d_in[0];
  const float* persona = (const float*)d_in[1];
  const int*   eindex  = (const int*)d_in[2];
  const float* gate_W  = (const float*)d_in[3];
  const float* gate_b  = (const float*)d_in[4];
  const float* Wp      = (const float*)d_in[5];
  const float* Wx      = (const float*)d_in[6];
  const float* att_W   = (const float*)d_in[7];
  const float* bias    = (const float*)d_in[8];
  int N = in_sizes[0] / CDIM;
  int E = in_sizes[2] / 2;
  const int* srcv = eindex;
  const int* dstv = eindex + E;
  float* out = (float*)d_out;

  // workspace layout (~110 MB): frags | asrc | adst | xf | mEnc | den | num
  char* ws = (char*)d_ws;
  __bf16* frags = (__bf16*)ws;
  size_t off = 128 * 1024;
  float*    asrc = (float*)(ws + off);    off += (size_t)HEADS * N * 4;
  float*    adst = (float*)(ws + off);    off += (size_t)HEADS * N * 4;
  float*    xf   = (float*)(ws + off);    off += (size_t)HEADS * N * HD * 4;
  unsigned* mEnc = (unsigned*)(ws + off); off += (size_t)HEADS * N * 4;
  float*    den  = (float*)(ws + off);    off += (size_t)HEADS * N * 4;
  float*    num  = (float*)(ws + off);    off += (size_t)HEADS * N * HD * 4;

  k_prep<<<dim3((2*FRAGS_PER_LAYER*FRAG_ELEMS + 255)/256), dim3(256), 0, stream>>>(
      gate_W, Wp, Wx, att_W, frags);

  long zwords = (long)HEADS * N * (2 + HD);   // mEnc+den+num are contiguous
  int  tiles  = (N + 15) / 16;
  int  b1     = (tiles + 7) / 8;
  long t2     = (long)(E + N) * HEADS;
  long t3     = t2 * 8;
  long t4     = (long)N * CDIM;

  for (int layer = 0; layer < 2; ++layer) {
    const float* xin = (layer == 0) ? x0 : out;
    k_fill0<<<dim3(2048), dim3(256), 0, stream>>>(mEnc, zwords);
    k1_node<<<dim3(b1), dim3(256), 0, stream>>>(xin, persona, frags, gate_b,
                                                asrc, adst, xf, N, layer);
    k2_max<<<dim3((int)((t2 + 255) / 256)), dim3(256), 0, stream>>>(
        srcv, dstv, asrc, adst, mEnc, N, E);
    k3_acc<<<dim3((int)((t3 + 255) / 256)), dim3(256), 0, stream>>>(
        srcv, dstv, asrc, adst, mEnc, den, xf, num, N, E);
    k4_fin<<<dim3((int)((t4 + 255) / 256)), dim3(256), 0, stream>>>(
        xin, num, den, bias, out, N, layer);
  }
  (void)n_in; (void)out_size; (void)ws_size;
}
